// Sch_33655363731905
// MI455X (gfx1250) — matrix operand order repaired and checked by reading, not compiled, not run on hardware
//
#include <hip/hip_runtime.h>

#define N_ATOMS   16384
#define E_EDGES   262144
#define HID       128
#define NFILT     128
#define NGAUSS    50
#define KG        64          // NGAUSS padded to WMMA K granularity
#define NINTER    6
#define LOG2F_    0.6931471805599453f

typedef __bf16 bf16;
typedef __attribute__((ext_vector_type(16))) __bf16 v16bf;
typedef __attribute__((ext_vector_type(8)))  float  v8f;
typedef __attribute__((ext_vector_type(8)))  __bf16 bf16x8;
typedef __attribute__((ext_vector_type(4)))  int    v4i;

// LDS row strides (elements), padded to dodge bank-conflict alignment
#define LDG  72    // for K=64 tiles
#define LDH  136   // for K=128 tiles

// ---- async global->LDS direct copy path (gfx1250), guarded fallback ------
#if defined(__has_builtin)
#if __has_builtin(__builtin_amdgcn_global_load_async_to_lds_b128) && \
    __has_builtin(__builtin_amdgcn_s_wait_asynccnt)
#define USE_ASYNC_LDS 1
#endif
#endif

#ifdef USE_ASYNC_LDS
__device__ __forceinline__ void async_copy16(const void* g, void* l) {
    __builtin_amdgcn_global_load_async_to_lds_b128(
        (__attribute__((address_space(1))) v4i*)(g),
        (__attribute__((address_space(3))) v4i*)(l), 0, 0);
}
__device__ __forceinline__ void async_wait0() {
    __builtin_amdgcn_s_wait_asynccnt(0);
}
#else
__device__ __forceinline__ void async_copy16(const void* g, void* l) {
    *(bf16x8*)l = *(const bf16x8*)g;
}
__device__ __forceinline__ void async_wait0() {}
#endif

__device__ __forceinline__ float sspf(float xv) {
    // shifted softplus: log(1+exp(x)) - log(2), numerically stable
    float ax = fabsf(xv);
    return fmaxf(xv, 0.0f) + log1pf(expf(-ax)) - LOG2F_;
}

// ---- WMMA fragment loaders (per CDNA5 ISA 7.12.2 VGPR layouts, wave32) ----
// A: 16x32 bf16 tile from row-major LDS. lane<16 holds K{0..7,16..23}, lane>=16 K{8..15,24..31}
__device__ __forceinline__ v16bf load_fragA(const bf16* base, int ld, int lane) {
    int m   = lane & 15;
    int khi = (lane >> 4) & 1;
    const bf16* row = base + m * ld + khi * 8;
    bf16x8 lo = *(const bf16x8*)(row);
    bf16x8 hi = *(const bf16x8*)(row + 16);
    v16bf a;
#pragma unroll
    for (int i = 0; i < 8; ++i) { a[i] = lo[i]; a[i + 8] = hi[i]; }
    return a;
}

// B: 32x16 bf16 tile read from TRANSPOSED weights WT[n][k] in LDS.
// lane holds column n=lane&15; lane<16 holds K{0..7,16..23}, lane>=16 K{8..15,24..31}
__device__ __forceinline__ v16bf load_fragB(const bf16* baseT, int ld, int lane) {
    int n  = lane & 15;
    int k0 = ((lane >> 4) & 1) * 8;
    const bf16* p = baseT + n * ld + k0;
    bf16x8 lo = *(const bf16x8*)(p);
    bf16x8 hi = *(const bf16x8*)(p + 16);
    v16bf b;
#pragma unroll
    for (int i = 0; i < 8; ++i) { b[i] = lo[i]; b[i + 8] = hi[i]; }
    return b;
}

// ---------------------------------------------------------------------------
// Edge geometry: d, Gaussian smear (bf16, padded to 64), cosine cutoff
__global__ void geom_kernel(const float* __restrict__ pos, const int* __restrict__ ei,
                            bf16* __restrict__ ea, float* __restrict__ C) {
    int e = blockIdx.x * blockDim.x + threadIdx.x;
    if (e >= E_EDGES) return;
    int s = ei[e], d = ei[E_EDGES + e];
    float dx = pos[s * 3 + 0] - pos[d * 3 + 0];
    float dy = pos[s * 3 + 1] - pos[d * 3 + 1];
    float dz = pos[s * 3 + 2] - pos[d * 3 + 2];
    float dist = sqrtf(dx * dx + dy * dy + dz * dz);
    C[e] = 0.5f * (cosf(dist * 0.31415926535897931f) + 1.0f);
    const float step  = 10.0f / 49.0f;
    const float coeff = -0.5f / (step * step);
    bf16* row = ea + (long)e * KG;
#pragma unroll 8
    for (int g = 0; g < KG; ++g) {
        float off = (float)g * step;
        float dd  = dist - off;
        float v   = (g < NGAUSS) ? expf(coeff * dd * dd) : 0.0f;
        row[g] = (bf16)v;
    }
}

// Convert [slices][K][128] f32 weights -> transposed padded bf16 [slices][128][Kpad]
__global__ void wconv_kernel(const float* __restrict__ src, bf16* __restrict__ dst,
                             int K, int Kpad, int total) {
    int idx = blockIdx.x * blockDim.x + threadIdx.x;
    if (idx >= total) return;
    int k  = idx % Kpad;
    int n  = (idx / Kpad) & 127;
    int sl = idx / (Kpad * 128);
    float v = (k < K) ? src[((long)sl * K + k) * 128 + n] : 0.0f;
    dst[idx] = (bf16)v;
}

// h = emb[z] ; keep f32 master + bf16 copy for WMMA
__global__ void inith_kernel(const int* __restrict__ z, const float* __restrict__ emb,
                             float* __restrict__ h, bf16* __restrict__ hb) {
    int idx = blockIdx.x * blockDim.x + threadIdx.x;
    if (idx >= N_ATOMS * HID) return;
    int n = idx >> 7, c = idx & 127;
    float v = emb[z[n] * HID + c];
    h[idx]  = v;
    hb[idx] = (bf16)v;
}

__global__ void zero_kernel(float* __restrict__ p, int n) {
    int idx = blockIdx.x * blockDim.x + threadIdx.x;
    if (idx < n) p[idx] = 0.0f;
}

// ---------------------------------------------------------------------------
// x = hb @ Wl1[i]   (N x 128) = (N x 128)(128 x 128), bf16 WMMA, f32 out
__global__ __launch_bounds__(256) void lin1_kernel(const bf16* __restrict__ hb,
                                                   const bf16* __restrict__ WT,
                                                   float* __restrict__ x) {
    __shared__ __align__(16) bf16 sH[128 * LDH];
    __shared__ __align__(16) bf16 sW[128 * LDH];
    const int tid  = threadIdx.x;
    const int lane = tid & 31;
    const int wv   = tid >> 5;
    const long rowBase = (long)blockIdx.x * 128;

    {   // cooperative copies: 128 rows x 128 bf16 each, direct to LDS
        int r = tid >> 1, hh = tid & 1;
        const bf16* gh = hb + (rowBase + r) * HID + hh * 64;
        const bf16* gw = WT + (long)r * 128 + hh * 64;
        bf16* dh = sH + r * LDH + hh * 64;
        bf16* dw = sW + r * LDH + hh * 64;
#pragma unroll
        for (int j = 0; j < 8; ++j) {
            async_copy16(gh + j * 8, dh + j * 8);
            async_copy16(gw + j * 8, dw + j * 8);
        }
    }
    async_wait0();
    __syncthreads();

    for (int t = 0; t < 8; ++t) {
        int tile = wv * 8 + t;
        int tr = tile >> 3, tc = tile & 7;
        v8f acc = {0.f, 0.f, 0.f, 0.f, 0.f, 0.f, 0.f, 0.f};
#pragma unroll
        for (int ks = 0; ks < 128; ks += 32) {
            v16bf a = load_fragA(sH + tr * 16 * LDH + ks, LDH, lane);
            v16bf b = load_fragB(sW + tc * 16 * LDH + ks, LDH, lane);
            acc = __builtin_amdgcn_wmma_f32_16x16x32_bf16(false, a, false, b, (short)0, acc, false, false);
        }
        int n  = tc * 16 + (lane & 15);
        int hi = (lane >> 4) & 1;
#pragma unroll
        for (int r = 0; r < 8; ++r) {
            int m = tr * 16 + r + 8 * hi;
            x[(rowBase + m) * NFILT + n] = acc[r];
        }
    }
}

// ---------------------------------------------------------------------------
// Fused edge pipeline for one interaction:
//   t1 = ssp(ea @ Wf1 + bf1); Wfilt = (t1 @ Wf2 + bf2) * C;
//   msg = x[src] * Wfilt; atomic agg[dst] += msg
__global__ __launch_bounds__(256) void edge_kernel(
    const bf16*  __restrict__ ea, const float* __restrict__ Cc,
    const int*   __restrict__ srcI, const int* __restrict__ dstI,
    const bf16*  __restrict__ W1T, const float* __restrict__ b1,
    const bf16*  __restrict__ W2T, const float* __restrict__ b2,
    const float* __restrict__ x,   float* __restrict__ agg) {
    __shared__ __align__(16) bf16 sEA[128 * LDG];
    __shared__ __align__(16) bf16 sW1[128 * LDG];
    __shared__ __align__(16) bf16 sT1[128 * LDH];
    __shared__ __align__(16) bf16 sW2[128 * LDH];
    __shared__ int   sSrc[128];
    __shared__ int   sDst[128];
    __shared__ float sC[128];

    const int tid  = threadIdx.x;
    const int lane = tid & 31;
    const int wv   = tid >> 5;
    const long eBase = (long)blockIdx.x * 128;

    {   // ea tile + W1T (128 x 64 bf16) and W2T (128 x 128 bf16), direct to LDS
        int r = tid >> 1, hh = tid & 1;
        const bf16* ge = ea  + (eBase + r) * KG + hh * 32;
        const bf16* gw = W1T + (long)r * KG + hh * 32;
        bf16* de = sEA + r * LDG + hh * 32;
        bf16* dw = sW1 + r * LDG + hh * 32;
#pragma unroll
        for (int j = 0; j < 4; ++j) {
            async_copy16(ge + j * 8, de + j * 8);
            async_copy16(gw + j * 8, dw + j * 8);
        }
        const bf16* g2 = W2T + (long)r * 128 + hh * 64;
        bf16* d2 = sW2 + r * LDH + hh * 64;
#pragma unroll
        for (int j = 0; j < 8; ++j) async_copy16(g2 + j * 8, d2 + j * 8);
        if (tid < 128) {
            sSrc[tid] = srcI[eBase + tid];
            sDst[tid] = dstI[eBase + tid];
            sC[tid]   = Cc[eBase + tid];
        }
    }
    async_wait0();
    __syncthreads();

    // GEMM1: [128x64]x[64x128] -> ssp -> sT1 (bf16)
    for (int t = 0; t < 8; ++t) {
        int tile = wv * 8 + t;
        int tr = tile >> 3, tc = tile & 7;
        v8f acc = {0.f, 0.f, 0.f, 0.f, 0.f, 0.f, 0.f, 0.f};
#pragma unroll
        for (int ks = 0; ks < KG; ks += 32) {
            v16bf a = load_fragA(sEA + tr * 16 * LDG + ks, LDG, lane);
            v16bf b = load_fragB(sW1 + tc * 16 * LDG + ks, LDG, lane);
            acc = __builtin_amdgcn_wmma_f32_16x16x32_bf16(false, a, false, b, (short)0, acc, false, false);
        }
        int n    = tc * 16 + (lane & 15);
        float bi = b1[n];
        int hi   = (lane >> 4) & 1;
#pragma unroll
        for (int r = 0; r < 8; ++r) {
            int m = tr * 16 + r + 8 * hi;
            sT1[m * LDH + n] = (bf16)sspf(acc[r] + bi);
        }
    }
    __syncthreads();

    // GEMM2 + cutoff + gather x[src] + scatter-add agg[dst]
    for (int t = 0; t < 8; ++t) {
        int tile = wv * 8 + t;
        int tr = tile >> 3, tc = tile & 7;
        v8f acc = {0.f, 0.f, 0.f, 0.f, 0.f, 0.f, 0.f, 0.f};
#pragma unroll
        for (int ks = 0; ks < 128; ks += 32) {
            v16bf a = load_fragA(sT1 + tr * 16 * LDH + ks, LDH, lane);
            v16bf b = load_fragB(sW2 + tc * 16 * LDH + ks, LDH, lane);
            acc = __builtin_amdgcn_wmma_f32_16x16x32_bf16(false, a, false, b, (short)0, acc, false, false);
        }
        int n    = tc * 16 + (lane & 15);
        float bi = b2[n];
        int hi   = (lane >> 4) & 1;
#pragma unroll
        for (int r = 0; r < 8; ++r) {
            int m = tr * 16 + r + 8 * hi;
            float wf  = (acc[r] + bi) * sC[m];
            int   s   = sSrc[m];
            int   dd  = sDst[m];
            float msg = wf * x[s * NFILT + n];
            atomicAdd(&agg[dd * NFILT + n], msg);
        }
    }
}

// ---------------------------------------------------------------------------
// x2 = ssp(agg @ Wl2 + bl2) @ Wl + bl ; h += x2 ; refresh bf16 copy
__global__ __launch_bounds__(256) void post_kernel(
    const float* __restrict__ agg,
    const bf16* __restrict__ W2T, const float* __restrict__ bl2,
    const bf16* __restrict__ WT,  const float* __restrict__ bl,
    float* __restrict__ h, bf16* __restrict__ hb) {
    __shared__ __align__(16) bf16 sA[128 * LDH];
    __shared__ __align__(16) bf16 sT[128 * LDH];
    __shared__ __align__(16) bf16 sW[128 * LDH];
    const int tid  = threadIdx.x;
    const int lane = tid & 31;
    const int wv   = tid >> 5;
    const long rowBase = (long)blockIdx.x * 128;

    {   // Wl2T via async path; agg f32 -> bf16 must convert through VALU
        int r = tid >> 1, hh = tid & 1;
        const bf16* gw = W2T + (long)r * 128 + hh * 64;
        bf16* dw = sW + r * LDH + hh * 64;
#pragma unroll
        for (int j = 0; j < 8; ++j) async_copy16(gw + j * 8, dw + j * 8);
        const float* ga = agg + (rowBase + r) * NFILT + hh * 64;
        bf16* da = sA + r * LDH + hh * 64;
#pragma unroll
        for (int j = 0; j < 64; ++j) da[j] = (bf16)ga[j];
    }
    async_wait0();
    __syncthreads();

    // GEMM1 -> ssp -> sT
    for (int t = 0; t < 8; ++t) {
        int tile = wv * 8 + t;
        int tr = tile >> 3, tc = tile & 7;
        v8f acc = {0.f, 0.f, 0.f, 0.f, 0.f, 0.f, 0.f, 0.f};
#pragma unroll
        for (int ks = 0; ks < 128; ks += 32) {
            v16bf a = load_fragA(sA + tr * 16 * LDH + ks, LDH, lane);
            v16bf b = load_fragB(sW + tc * 16 * LDH + ks, LDH, lane);
            acc = __builtin_amdgcn_wmma_f32_16x16x32_bf16(false, a, false, b, (short)0, acc, false, false);
        }
        int n    = tc * 16 + (lane & 15);
        float bi = bl2[n];
        int hi   = (lane >> 4) & 1;
#pragma unroll
        for (int r = 0; r < 8; ++r) {
            int m = tr * 16 + r + 8 * hi;
            sT[m * LDH + n] = (bf16)sspf(acc[r] + bi);
        }
    }
    __syncthreads();
    {   // overwrite sW with WlT (all prior readers passed the barrier)
        int r = tid >> 1, hh = tid & 1;
        const bf16* gw = WT + (long)r * 128 + hh * 64;
        bf16* dw = sW + r * LDH + hh * 64;
#pragma unroll
        for (int j = 0; j < 8; ++j) async_copy16(gw + j * 8, dw + j * 8);
    }
    async_wait0();
    __syncthreads();

    // GEMM2 -> residual add into h
    for (int t = 0; t < 8; ++t) {
        int tile = wv * 8 + t;
        int tr = tile >> 3, tc = tile & 7;
        v8f acc = {0.f, 0.f, 0.f, 0.f, 0.f, 0.f, 0.f, 0.f};
#pragma unroll
        for (int ks = 0; ks < 128; ks += 32) {
            v16bf a = load_fragA(sT + tr * 16 * LDH + ks, LDH, lane);
            v16bf b = load_fragB(sW + tc * 16 * LDH + ks, LDH, lane);
            acc = __builtin_amdgcn_wmma_f32_16x16x32_bf16(false, a, false, b, (short)0, acc, false, false);
        }
        int n    = tc * 16 + (lane & 15);
        float bi = bl[n];
        int hi   = (lane >> 4) & 1;
#pragma unroll
        for (int r = 0; r < 8; ++r) {
            int m   = tr * 16 + r + 8 * hi;
            long idx = (rowBase + m) * HID + n;
            float nh = h[idx] + acc[r] + bi;
            h[idx]  = nh;
            hb[idx] = (bf16)nh;
        }
    }
}

// ---------------------------------------------------------------------------
// readout: sum_n ( ssp(h[n] @ Wo1 + bo1) @ Wo2 + bo2 ), then relu
__global__ __launch_bounds__(64) void out_kernel(const float* __restrict__ h,
                                                 const float* __restrict__ Wo1,
                                                 const float* __restrict__ bo1,
                                                 const float* __restrict__ Wo2,
                                                 const float* __restrict__ bo2,
                                                 float* __restrict__ acc) {
    __shared__ float sh[128];
    __shared__ float sp[64];
    int node = blockIdx.x;
    int j = threadIdx.x;
    sh[j]      = h[(long)node * HID + j];
    sh[j + 64] = h[(long)node * HID + 64 + j];
    __syncthreads();
    float a = bo1[j];
#pragma unroll 8
    for (int k = 0; k < 128; ++k) a += sh[k] * Wo1[k * 64 + j];
    sp[j] = sspf(a) * Wo2[j];
    __syncthreads();
    if (j == 0) {
        float s = bo2[0];
#pragma unroll
        for (int k = 0; k < 64; ++k) s += sp[k];
        atomicAdd(acc, s);
    }
}

__global__ void finish_kernel(const float* __restrict__ acc, float* __restrict__ out) {
    if (threadIdx.x == 0) out[0] = fmaxf(acc[0], 0.0f);
}

// ---------------------------------------------------------------------------
extern "C" void kernel_launch(void* const* d_in, const int* in_sizes, int n_in,
                              void* d_out, int out_size, void* d_ws, size_t ws_size,
                              hipStream_t stream) {
    const int*   z    = (const int*)  d_in[0];
    const float* pos  = (const float*)d_in[1];
    const int*   ei   = (const int*)  d_in[2];
    const float* emb  = (const float*)d_in[3];
    const float* Wf1  = (const float*)d_in[4];
    const float* bf1  = (const float*)d_in[5];
    const float* Wf2  = (const float*)d_in[6];
    const float* bf2  = (const float*)d_in[7];
    const float* Wl1  = (const float*)d_in[8];
    const float* Wl2  = (const float*)d_in[9];
    const float* bl2  = (const float*)d_in[10];
    const float* Wl   = (const float*)d_in[11];
    const float* bl   = (const float*)d_in[12];
    const float* Wo1  = (const float*)d_in[13];
    const float* bo1  = (const float*)d_in[14];
    const float* Wo2  = (const float*)d_in[15];
    const float* bo2  = (const float*)d_in[16];

    char* w = (char*)d_ws;
    bf16*  ea   = (bf16*)w;  w += (size_t)E_EDGES * KG * 2;
    float* C    = (float*)w; w += (size_t)E_EDGES * 4;
    float* x    = (float*)w; w += (size_t)N_ATOMS * NFILT * 4;
    float* agg  = (float*)w; w += (size_t)N_ATOMS * NFILT * 4;
    float* h    = (float*)w; w += (size_t)N_ATOMS * HID * 4;
    bf16*  hb   = (bf16*)w;  w += (size_t)N_ATOMS * HID * 2;
    bf16*  W1T  = (bf16*)w;  w += (size_t)NINTER * 128 * KG * 2;
    bf16*  W2T  = (bf16*)w;  w += (size_t)NINTER * 128 * 128 * 2;
    bf16*  Wl1T = (bf16*)w;  w += (size_t)NINTER * 128 * 128 * 2;
    bf16*  Wl2T = (bf16*)w;  w += (size_t)NINTER * 128 * 128 * 2;
    bf16*  WlT  = (bf16*)w;  w += (size_t)NINTER * 128 * 128 * 2;
    float* acc  = (float*)w; w += 256;

    // preprocessing
    geom_kernel<<<E_EDGES / 256, 256, 0, stream>>>(pos, ei, ea, C);
    {
        int t1 = NINTER * 128 * KG;
        int t2 = NINTER * 128 * 128;
        wconv_kernel<<<(t1 + 255) / 256, 256, 0, stream>>>(Wf1, W1T, NGAUSS, KG, t1);
        wconv_kernel<<<(t2 + 255) / 256, 256, 0, stream>>>(Wf2, W2T, 128, 128, t2);
        wconv_kernel<<<(t2 + 255) / 256, 256, 0, stream>>>(Wl1, Wl1T, 128, 128, t2);
        wconv_kernel<<<(t2 + 255) / 256, 256, 0, stream>>>(Wl2, Wl2T, 128, 128, t2);
        wconv_kernel<<<(t2 + 255) / 256, 256, 0, stream>>>(Wl,  WlT,  128, 128, t2);
    }
    inith_kernel<<<(N_ATOMS * HID) / 256, 256, 0, stream>>>(z, emb, h, hb);

    for (int i = 0; i < NINTER; ++i) {
        lin1_kernel<<<N_ATOMS / 128, 256, 0, stream>>>(hb, Wl1T + (size_t)i * 128 * 128, x);
        zero_kernel<<<(N_ATOMS * NFILT) / 256, 256, 0, stream>>>(agg, N_ATOMS * NFILT);
        edge_kernel<<<E_EDGES / 128, 256, 0, stream>>>(
            ea, C, ei, ei + E_EDGES,
            W1T + (size_t)i * 128 * KG, bf1 + i * NFILT,
            W2T + (size_t)i * 128 * 128, bf2 + i * NFILT,
            x, agg);
        post_kernel<<<N_ATOMS / 128, 256, 0, stream>>>(
            agg,
            Wl2T + (size_t)i * 128 * 128, bl2 + i * HID,
            WlT  + (size_t)i * 128 * 128, bl  + i * HID,
            h, hb);
    }

    zero_kernel<<<1, 32, 0, stream>>>(acc, 1);
    out_kernel<<<N_ATOMS, 64, 0, stream>>>(h, Wo1, bo1, Wo2, bo2, acc);
    finish_kernel<<<1, 32, 0, stream>>>(acc, (float*)d_out);
}
